// Mambalayer_fea_down_20770461843913
// MI455X (gfx1250) — matrix operand order repaired and checked by reading, not compiled, not run on hardware
//
#include <hip/hip_runtime.h>

// ---- model dims ----
#define BATCH   4
#define DMODEL  512
#define LSEQ    1024
#define DSTATE  16
#define DCONV   4
#define DINNER  1024
#define DTRANK  32
#define NTOK    (BATCH * LSEQ)      // 4096 rows for all GEMMs

typedef __attribute__((ext_vector_type(16))) __bf16 v16bf;
typedef __attribute__((ext_vector_type(8)))  __bf16 v8bf;
typedef __attribute__((ext_vector_type(8)))  float  v8f;
typedef __attribute__((ext_vector_type(4)))  unsigned int u32x4;
typedef __attribute__((ext_vector_type(8)))  int i32x8;
typedef __attribute__((ext_vector_type(4)))  int i32x4;

#if defined(__HIP_DEVICE_COMPILE__) && __has_builtin(__builtin_amdgcn_tensor_load_to_lds)
#define HAVE_TDM 1
#else
#define HAVE_TDM 0
#endif

// round-to-nearest-even f32 -> bf16
__device__ __forceinline__ __bf16 f2bf(float f) {
  unsigned int u = __float_as_uint(f);
  unsigned int r = (u + 0x7FFFu + ((u >> 16) & 1u)) >> 16;
  unsigned short s = (unsigned short)r;
  return *reinterpret_cast<__bf16*>(&s);
}

__device__ __forceinline__ float sigmoidf_dev(float x) {
  return 1.0f / (1.0f + __expf(-x));
}

// ---------------- elementwise kernels ----------------

__global__ void k_f32_to_bf16(const float* __restrict__ in, __bf16* __restrict__ out, int n) {
  int i = blockIdx.x * blockDim.x + threadIdx.x;
  if (i < n) out[i] = f2bf(in[i]);
}

// x [B, DMODEL, L] (f32) -> xt [B*L, DMODEL] (bf16)
__global__ void k_transpose_x(const float* __restrict__ x, __bf16* __restrict__ xt, int n) {
  int o = blockIdx.x * blockDim.x + threadIdx.x;
  if (o >= n) return;
  int d = o % DMODEL;
  int r = o / DMODEL;        // b*L + l
  int l = r % LSEQ;
  int b = r / LSEQ;
  xt[o] = f2bf(x[((size_t)(b * DMODEL + d)) * LSEQ + l]);
}

// depthwise causal conv + bias + SiLU. x_in = xz[:, 0:DINNER], row stride 2*DINNER.
__global__ void k_conv_silu(const float* __restrict__ xz,
                            const float* __restrict__ conv_w, const float* __restrict__ conv_b,
                            float* __restrict__ xc, __bf16* __restrict__ xc_bf, int n) {
  int o = blockIdx.x * blockDim.x + threadIdx.x;
  if (o >= n) return;
  int e = o % DINNER;
  int r = o / DINNER;        // b*L + l
  int l = r % LSEQ;
  int b = r / LSEQ;
  float acc = conv_b[e];
#pragma unroll
  for (int j = 0; j < DCONV; ++j) {
    int lp = l - (DCONV - 1) + j;
    if (lp >= 0) {
      float xv = xz[((size_t)(b * LSEQ + lp)) * (2 * DINNER) + e];
      acc += conv_w[e * DCONV + j] * xv;
    }
  }
  float v = acc * sigmoidf_dev(acc);   // SiLU
  xc[o] = v;
  xc_bf[o] = f2bf(v);
}

// dbl[:, 0:DTRANK] (f32) -> dt_in (bf16) [NTOK, DTRANK]
__global__ void k_dt_slice(const float* __restrict__ dbl, __bf16* __restrict__ dt_in, int n) {
  int o = blockIdx.x * blockDim.x + threadIdx.x;
  if (o >= n) return;
  int rr = o % DTRANK;
  int row = o / DTRANK;
  dt_in[o] = f2bf(dbl[(size_t)row * (DTRANK + 2 * DSTATE) + rr]);
}

// ---------------- selective scan ----------------
__global__ __launch_bounds__(256) void k_scan(
    const float* __restrict__ dt_lin,   // [NTOK, DINNER]
    const float* __restrict__ dt_b,     // [DINNER]
    const float* __restrict__ dbl,      // [NTOK, 64]  (dt | B | C)
    const float* __restrict__ xc,       // [NTOK, DINNER]
    const float* __restrict__ xz,       // [NTOK, 2*DINNER] (z = cols DINNER..)
    const float* __restrict__ A_log,    // [DINNER, DSTATE]
    const float* __restrict__ Dvec,     // [DINNER]
    __bf16* __restrict__ ybf)           // [NTOK, DINNER]
{
  int e = blockIdx.x * blockDim.x + threadIdx.x;
  int b = blockIdx.y;
  if (e >= DINNER) return;

  float Arow[DSTATE];
#pragma unroll
  for (int n = 0; n < DSTATE; ++n) Arow[n] = -__expf(A_log[e * DSTATE + n]);

  float h[DSTATE];
#pragma unroll
  for (int n = 0; n < DSTATE; ++n) h[n] = 0.0f;

  const float dpb = dt_b[e];
  const float Dv  = Dvec[e];

  for (int l = 0; l < LSEQ; ++l) {
    size_t row = (size_t)b * LSEQ + l;
    float dtl = dt_lin[row * DINNER + e] + dpb;
    float dt  = (dtl > 20.0f) ? dtl : log1pf(__expf(dtl));   // softplus
    float xv  = xc[row * DINNER + e];
    const float* dr = dbl + row * (DTRANK + 2 * DSTATE);
    float dtx = dt * xv;
    float y = 0.0f;
#pragma unroll
    for (int n = 0; n < DSTATE; ++n) {
      float dA = __expf(dt * Arow[n]);
      h[n] = dA * h[n] + dtx * dr[DTRANK + n];         // B
      y   += h[n] * dr[DTRANK + DSTATE + n];           // C
    }
    float zv = xz[row * (2 * DINNER) + DINNER + e];
    float yo = (y + xv * Dv) * (zv * sigmoidf_dev(zv));
    ybf[row * DINNER + e] = f2bf(yo);
  }
}

// ---------------- TDM helpers ----------------
#define KCHUNK  64
#define LDS_ROW 144   // 128B of data + 16B TDM pad per row -> conflict-free fragment reads

#if HAVE_TDM
// Issue a TDM load of a 64-row x 64-elem bf16 tile (row stride K elems) into LDS.
// D# per ISA 08: group0 {count, lds_addr, global_addr, type=2};
// group1 {data_size=2B, pad every 32 DWORDs by 4 DWORDs, dims/strides}.
// This toolchain's builtin is the 6-arg form: (g0, g1, g2, g3, g_extra, cpol).
__device__ __forceinline__ void tdm_tile(const __bf16* g, unsigned lds_off, int K) {
  unsigned long long ga = (unsigned long long)(uintptr_t)g;
  u32x4 g0;
  g0[0] = 1u;                                              // count=1
  g0[1] = lds_off;                                         // LDS byte address
  g0[2] = (unsigned)(ga & 0xFFFFFFFFu);                    // global_addr[31:0]
  g0[3] = (unsigned)((ga >> 32) & 0x01FFFFFFu) | (2u << 30); // addr[56:32] | type=2
  i32x8 g1;
  g1[0] = (1 << 16) | (1 << 20) | (4 << 22) | (3 << 25);   // 2B elems, pad_en, ivl=32dw, amt=4dw
  g1[1] = (int)((unsigned)(K & 0xFFFF) << 16);             // tensor_dim0[15:0]
  g1[2] = (int)(((unsigned)K >> 16) & 0xFFFFu) | (64u << 16); // dim0[31:16] | tensor_dim1=64
  g1[3] = (int)(64u << 16);                                // tile_dim0 = 64
  g1[4] = 64;                                              // tile_dim1 = 64, tile_dim2 = 0
  g1[5] = K;                                               // tensor_dim0_stride[31:0]
  g1[6] = 0;                                               // stride[47:32] | dim1_stride lo
  g1[7] = 0;
  i32x4 z4 = {0, 0, 0, 0};
  i32x8 z8 = {0, 0, 0, 0, 0, 0, 0, 0};
  __builtin_amdgcn_tensor_load_to_lds(g0, g1, z4, z4, z8, 0);
}
#endif

// Stage the A (64 x KCHUNK) and W (64 x KCHUNK) tiles for chunk at gA/gW into LDS.
__device__ __forceinline__ void stage_tiles(unsigned char* dA, const __bf16* gA,
                                            unsigned char* dW, const __bf16* gW, int K) {
#if HAVE_TDM
  if (threadIdx.x < 32) {   // one wave issues the DMA; TDM ignores EXEC, branch is wave-uniform
    tdm_tile(gA, (unsigned)(uintptr_t)dA, K);
    tdm_tile(gW, (unsigned)(uintptr_t)dW, K);
  }
#else
  const unsigned short* sa = (const unsigned short*)gA;
  const unsigned short* sw = (const unsigned short*)gW;
  for (int i = threadIdx.x; i < 64 * KCHUNK; i += 128) {
    int r = i >> 6, k = i & (KCHUNK - 1);
    *(unsigned short*)(dA + r * LDS_ROW + k * 2) = sa[(size_t)r * K + k];
    *(unsigned short*)(dW + r * LDS_ROW + k * 2) = sw[(size_t)r * K + k];
  }
#endif
}

__device__ __forceinline__ void stage_wait(bool more) {
#if HAVE_TDM
  if (threadIdx.x < 32) {
    if (more) __builtin_amdgcn_s_wait_tensorcnt(2);   // prev chunk done (in-order)
    else      __builtin_amdgcn_s_wait_tensorcnt(0);
  }
#endif
}

// ---------------- TDM double-buffered bf16 WMMA GEMM ----------------
// C[M,N] (f32) = A[M,K] x W[N,K]^T, K % 64 == 0. Block = 4 waves = 64x64 tile;
// A and W tiles staged in LDS by the Tensor Data Mover, WMMAs fed by ds loads.
__global__ __launch_bounds__(128) void k_gemm_tdm(
    const __bf16* __restrict__ A, const __bf16* __restrict__ W,
    float* __restrict__ C, int M, int N, int K)
{
  __shared__ __align__(16) unsigned char smA[2][64 * LDS_ROW];
  __shared__ __align__(16) unsigned char smW[2][64 * LDS_ROW];

  int lane = threadIdx.x & 31;
  int wave = threadIdx.x >> 5;
  int m0 = blockIdx.y * 64;
  int n0 = blockIdx.x * 64;
  int nchunks = K / KCHUNK;

  v8f acc[4] = {};

  const __bf16* Abase = A + (size_t)m0 * K;
  const __bf16* Wbase = W + (size_t)n0 * K;

  stage_tiles(smA[0], Abase, smW[0], Wbase, K);   // prologue: chunk 0

  int half = lane >> 4;                               // lane group 0/1
  unsigned aOff = (unsigned)((wave * 16 + (lane & 15)) * LDS_ROW + half * 16);
  unsigned wCol = (unsigned)((lane & 15) * LDS_ROW + half * 16);

  for (int c = 0; c < nchunks; ++c) {
    if (c + 1 < nchunks) {
      stage_tiles(smA[(c + 1) & 1], Abase + (c + 1) * KCHUNK,
                  smW[(c + 1) & 1], Wbase + (c + 1) * KCHUNK, K);
      stage_wait(true);
    } else {
      stage_wait(false);
    }
    __syncthreads();                                  // chunk c resident

    const unsigned char* bufA = smA[c & 1];
    const unsigned char* bufW = smW[c & 1];
#pragma unroll
    for (int s = 0; s < 2; ++s) {                     // two 32-K WMMA steps per chunk
      v8bf lo = *(const v8bf*)(bufA + aOff + s * 64);
      v8bf hi = *(const v8bf*)(bufA + aOff + s * 64 + 32);
      v16bf a;
#pragma unroll
      for (int i = 0; i < 8; ++i) { a[i] = lo[i]; a[i + 8] = hi[i]; }
#pragma unroll
      for (int j = 0; j < 4; ++j) {
        const unsigned char* wp = bufW + wCol + (unsigned)(j * 16 * LDS_ROW) + s * 64;
        v8bf b0 = *(const v8bf*)(wp);
        v8bf b1 = *(const v8bf*)(wp + 32);
        v16bf bf;
#pragma unroll
        for (int i = 0; i < 8; ++i) { bf[i] = b0[i]; bf[i + 8] = b1[i]; }
        acc[j] = __builtin_amdgcn_wmma_f32_16x16x32_bf16(
            false, a, false, bf, (short)0, acc[j], false, false);
      }
    }
    __syncthreads();                                  // buffer safe to overwrite
  }

  int rbase = m0 + wave * 16 + (half << 3);
  int col0  = lane & 15;
#pragma unroll
  for (int j = 0; j < 4; ++j) {
    int col = n0 + j * 16 + col0;
#pragma unroll
    for (int r = 0; r < 8; ++r)
      C[(size_t)(rbase + r) * N + col] = acc[j][r];
  }
}

// ---------------- direct-global bf16 WMMA GEMM (small K, e.g. dt_proj K=32) ----------------
__global__ __launch_bounds__(128) void k_gemm_direct(
    const __bf16* __restrict__ A, const __bf16* __restrict__ W,
    float* __restrict__ C, int M, int N, int K)
{
  int lane = threadIdx.x & 31;
  int wave = threadIdx.x >> 5;
  int m0 = (blockIdx.y * 4 + wave) * 16;
  int n0 = blockIdx.x * 64;
  if (m0 >= M || n0 >= N) return;   // wave-uniform

  v8f acc[4] = {};

  int arow  = m0 + (lane & 15);
  int koffA = (lane >> 4) * 16;
  int bcol  = n0 + (lane & 15);
  int kbB   = (lane >> 4) * 16;

  const __bf16* Ap = A + (size_t)arow * K;

  for (int k0 = 0; k0 < K; k0 += 32) {
    v8bf lo = *(const v8bf*)(Ap + k0 + koffA);
    v8bf hi = *(const v8bf*)(Ap + k0 + koffA + 8);
    v16bf a;
#pragma unroll
    for (int i = 0; i < 8; ++i) { a[i] = lo[i]; a[i + 8] = hi[i]; }
#pragma unroll
    for (int j = 0; j < 4; ++j) {
      const __bf16* Wp = W + (size_t)(bcol + j * 16) * K + k0 + kbB;
      v16bf bf = *(const v16bf*)Wp;
      acc[j] = __builtin_amdgcn_wmma_f32_16x16x32_bf16(
          false, a, false, bf, (short)0, acc[j], false, false);
    }
  }

  int rbase = m0 + ((lane >> 4) << 3);
  int col0  = lane & 15;
#pragma unroll
  for (int j = 0; j < 4; ++j) {
    int col = n0 + j * 16 + col0;
#pragma unroll
    for (int r = 0; r < 8; ++r)
      C[(size_t)(rbase + r) * N + col] = acc[j][r];
  }
}

// out_tmp [B*L, DMODEL] (f32) -> d_out [B, DMODEL, L] (f32)
__global__ void k_out_transpose(const float* __restrict__ t, float* __restrict__ out, int n) {
  int o = blockIdx.x * blockDim.x + threadIdx.x;
  if (o >= n) return;
  int l = o % LSEQ;
  int r = o / LSEQ;          // b*DMODEL + d
  int d = r % DMODEL;
  int b = r / DMODEL;
  out[o] = t[((size_t)(b * LSEQ + l)) * DMODEL + d];
}

// ---------------- host side ----------------

static inline size_t align256(size_t x) { return (x + 255) & ~(size_t)255; }

extern "C" void kernel_launch(void* const* d_in, const int* in_sizes, int n_in,
                              void* d_out, int out_size, void* d_ws, size_t ws_size,
                              hipStream_t stream) {
  const float* x        = (const float*)d_in[0];
  const float* in_w     = (const float*)d_in[1];   // [2*DINNER, DMODEL]
  const float* conv_w   = (const float*)d_in[2];   // [DINNER, 1, DCONV]
  const float* conv_b   = (const float*)d_in[3];
  const float* xproj_w  = (const float*)d_in[4];   // [64, DINNER]
  const float* dtproj_w = (const float*)d_in[5];   // [DINNER, DTRANK]
  const float* dtproj_b = (const float*)d_in[6];
  const float* A_log    = (const float*)d_in[7];
  const float* Dvec     = (const float*)d_in[8];
  const float* out_w    = (const float*)d_in[9];   // [DMODEL, DINNER]
  float* outp = (float*)d_out;

  // workspace bump allocator
  char* ws = (char*)d_ws;
  size_t off = 0;
  auto take = [&](size_t bytes) { char* p = ws + off; off += align256(bytes); return p; };

  __bf16* xt_bf   = (__bf16*)take((size_t)NTOK * DMODEL * 2);
  __bf16* win_bf  = (__bf16*)take((size_t)2 * DINNER * DMODEL * 2);
  __bf16* wxp_bf  = (__bf16*)take((size_t)(DTRANK + 2 * DSTATE) * DINNER * 2);
  __bf16* wdt_bf  = (__bf16*)take((size_t)DINNER * DTRANK * 2);
  __bf16* wout_bf = (__bf16*)take((size_t)DMODEL * DINNER * 2);
  float*  xz      = (float*)take((size_t)NTOK * 2 * DINNER * 4);
  float*  xc_f    = (float*)take((size_t)NTOK * DINNER * 4);
  __bf16* xc_bf   = (__bf16*)take((size_t)NTOK * DINNER * 2);
  float*  dbl     = (float*)take((size_t)NTOK * (DTRANK + 2 * DSTATE) * 4);
  __bf16* dt_in   = (__bf16*)take((size_t)NTOK * DTRANK * 2);
  float*  dt_lin  = (float*)take((size_t)NTOK * DINNER * 4);
  __bf16* y_bf    = (__bf16*)take((size_t)NTOK * DINNER * 2);
  float*  out_tmp = (float*)take((size_t)NTOK * DMODEL * 4);
  (void)ws_size; (void)n_in; (void)in_sizes; (void)out_size;

  const int TB = 256;
  auto blocks = [](int n, int tb) { return (n + tb - 1) / tb; };

  // weights -> bf16
  int n_in_w  = 2 * DINNER * DMODEL;
  int n_xp_w  = (DTRANK + 2 * DSTATE) * DINNER;
  int n_dt_w  = DINNER * DTRANK;
  int n_out_w = DMODEL * DINNER;
  k_f32_to_bf16<<<blocks(n_in_w, TB),  TB, 0, stream>>>(in_w,     win_bf,  n_in_w);
  k_f32_to_bf16<<<blocks(n_xp_w, TB),  TB, 0, stream>>>(xproj_w,  wxp_bf,  n_xp_w);
  k_f32_to_bf16<<<blocks(n_dt_w, TB),  TB, 0, stream>>>(dtproj_w, wdt_bf,  n_dt_w);
  k_f32_to_bf16<<<blocks(n_out_w, TB), TB, 0, stream>>>(out_w,    wout_bf, n_out_w);

  // x -> xt (bf16, [B*L, DMODEL])
  int n_xt = NTOK * DMODEL;
  k_transpose_x<<<blocks(n_xt, TB), TB, 0, stream>>>(x, xt_bf, n_xt);

  // in_proj: xz = xt @ in_w^T   [4096, 2048], K=512  (TDM-staged)
  k_gemm_tdm<<<dim3((2 * DINNER) / 64, NTOK / 64), 128, 0, stream>>>(
      xt_bf, win_bf, xz, NTOK, 2 * DINNER, DMODEL);

  // depthwise causal conv + SiLU
  int n_xc = NTOK * DINNER;
  k_conv_silu<<<blocks(n_xc, TB), TB, 0, stream>>>(xz, conv_w, conv_b, xc_f, xc_bf, n_xc);

  // x_proj: dbl = xc @ xproj_w^T   [4096, 64], K=1024  (TDM-staged)
  k_gemm_tdm<<<dim3((DTRANK + 2 * DSTATE) / 64, NTOK / 64), 128, 0, stream>>>(
      xc_bf, wxp_bf, dbl, NTOK, DTRANK + 2 * DSTATE, DINNER);

  // dt slice -> bf16
  int n_dt = NTOK * DTRANK;
  k_dt_slice<<<blocks(n_dt, TB), TB, 0, stream>>>(dbl, dt_in, n_dt);

  // dt_proj: dt_lin = dt_in @ dtproj_w^T   [4096, 1024], K=32  (direct)
  k_gemm_direct<<<dim3(DINNER / 64, NTOK / 64), 128, 0, stream>>>(
      dt_in, wdt_bf, dt_lin, NTOK, DINNER, DTRANK);

  // selective scan + D-skip + SiLU(z) gate -> y_bf
  k_scan<<<dim3(DINNER / 256, BATCH), 256, 0, stream>>>(
      dt_lin, dtproj_b, dbl, xc_f, xz, A_log, Dvec, y_bf);

  // out_proj: out_tmp = y @ out_w^T   [4096, 512], K=1024  (TDM-staged)
  k_gemm_tdm<<<dim3(DMODEL / 64, NTOK / 64), 128, 0, stream>>>(
      y_bf, wout_bf, out_tmp, NTOK, DMODEL, DINNER);

  // final transpose to [B, DMODEL, L]
  int n_out = NTOK * DMODEL;
  k_out_transpose<<<blocks(n_out, TB), TB, 0, stream>>>(out_tmp, outp, n_out);
}
